// Transformer_34179349741885
// MI455X (gfx1250) — matrix operand order repaired and checked by reading, not compiled, not run on hardware
//
#include <hip/hip_runtime.h>
#include <math.h>

// ---------------------------------------------------------------------------
// Transformer forward for gfx1250 (MI455X).
//  - One templated WMMA GEMM (v_wmma_f32_16x16x32_f16, wave tile 32x32,
//    BK=64 -> 8 WMMAs per wave per barrier pair) serves projections, FFN,
//    batched attention scores (QK^T) and batched prob@V.
//  - f16 operands in global memory (softmax probs, f16 embed table) are
//    staged with global_load_async_to_lds_b128 + s_wait_asynccnt (ASYNCcnt).
// ---------------------------------------------------------------------------

typedef __attribute__((ext_vector_type(16))) _Float16 v16h;
typedef __attribute__((ext_vector_type(8)))  float    v8f;

#define XL  6
#define XD  512
#define XH  2048
#define XNH 8
#define XHD 64
#define XV  32000
#define XSEQ 512
#define XB  4
#define XT  (XSEQ * XB)      // 2048 tokens
#define NEG_INF (-100000.0f)

#define BK 64
#define LDSA (BK + 8)        // 72 halfs => 144 B row stride (16B aligned)
#define LDSB (BK + 8)

// fragment loaders (A 16x32 / B 32x16 wave32, contiguous k = 16*hi + i in both A and B)
__device__ __forceinline__ v16h load_afrag(const _Float16* As, int row, int ks,
                                           int lrow, int hi)
{
    v16h a;
    const _Float16* ap = As + (row + lrow) * LDSA + ks + hi * 16;
    #pragma unroll
    for (int i = 0; i < 16; ++i) a[i] = ap[i];
    return a;
}
__device__ __forceinline__ v16h load_bfrag(const _Float16* Bs, int col, int ks,
                                           int lrow, int hi)
{
    v16h b;
    const _Float16* bp = Bs + (col + lrow) * LDSB + ks + hi * 16;
    #pragma unroll
    for (int i = 0; i < 16; ++i) b[i] = bp[i];
    return b;
}

// ===================== generic batched WMMA GEMM ===========================
// C = A @ B + bias (+residual)(+relu), strided/batched.
//  TBM x TBN block tile (TBM/32 * TBN/32 == 8 waves), BK=64.
//  BT : B is [N,K] with leading dim ldb (else [K,N])
//  AH : A is f16 (staged via async global->LDS copy), else f32 (cvt staging)
//  CH : C is f16, else f32
template<int TBM, int TBN, int BT, int AH, int CH>
__global__ __launch_bounds__(256)
void gemm2_kernel(const void* __restrict__ Ap, const float* __restrict__ Bp,
                  const float* __restrict__ bias,
                  const float* __restrict__ residual,
                  void* __restrict__ Cp,
                  int M, int N, int K, int lda, int ldb, int ldc,
                  long sA, long sB, long sC, int do_relu)
{
    __shared__ _Float16 As[TBM * LDSA];
    __shared__ _Float16 Bs[TBN * LDSB];

    const int batch = blockIdx.z;
    const float*    A   = (const float*)Ap    + (AH ? 0 : (size_t)batch * sA);
    const _Float16* A16 = (const _Float16*)Ap + (AH ? (size_t)batch * sA : 0);
    const float*    B   = Bp + (size_t)batch * sB;
    float*    C   = (float*)Cp    + (CH ? 0 : (size_t)batch * sC);
    _Float16* C16 = (_Float16*)Cp + (CH ? (size_t)batch * sC : 0);

    const int tid  = threadIdx.x;
    const int lane = tid & 31;
    const int wave = tid >> 5;
    const int wm   = wave & (TBM / 32 - 1);
    const int wn   = wave / (TBM / 32);
    const int m0   = blockIdx.y * TBM;
    const int n0   = blockIdx.x * TBN;
    const int lrow = lane & 15;
    const int hi   = lane >> 4;

    v8f c00 = {}, c01 = {}, c10 = {}, c11 = {};

    for (int k0 = 0; k0 < K; k0 += BK) {
        // ---- stage A (TBM x 64) ------------------------------------------
        #pragma unroll
        for (int c = tid; c < TBM * 4; c += 256) {
            const int r  = c >> 2;
            const int kc = (c & 3) * 16;
            _Float16* sa = As + r * LDSA + kc;
            if constexpr (AH) {
                const _Float16* ga = A16 + (size_t)(m0 + r) * lda + k0 + kc;
                const unsigned int lo = (unsigned int)(uintptr_t)sa; // LDS off = addr[31:0]
                asm volatile("global_load_async_to_lds_b128 %0, %1, off"
                             :: "v"(lo), "v"(ga) : "memory");
                asm volatile("global_load_async_to_lds_b128 %0, %1, off"
                             :: "v"(lo + 16u), "v"(ga + 8) : "memory");
            } else {
                const float* ga = A + (size_t)(m0 + r) * lda + k0 + kc;
                #pragma unroll
                for (int i = 0; i < 16; ++i) sa[i] = (_Float16)ga[i];
                if (k0 + BK < K) __builtin_prefetch(ga + BK, 0, 0);
            }
        }
        // ---- stage B into Bs[n][k] ---------------------------------------
        if constexpr (BT) {          // B[N,K]: contiguous copy + cvt
            #pragma unroll
            for (int c = tid; c < TBN * 4; c += 256) {
                const int nn = c >> 2;
                const int kc = (c & 3) * 16;
                const float* gb = B + (size_t)(n0 + nn) * ldb + k0 + kc;
                _Float16* sb = Bs + nn * LDSB + kc;
                #pragma unroll
                for (int i = 0; i < 16; ++i) sb[i] = (_Float16)gb[i];
            }
        } else {                     // B[K,N]: coalesced read, scatter to LDS
            constexpr int NPT = TBN / 8;
            const int nc = (tid % NPT) * 8;
            #pragma unroll
            for (int kk = tid / NPT; kk < BK; kk += 256 / NPT) {
                const float* gb = B + (size_t)(k0 + kk) * ldb + n0 + nc;
                #pragma unroll
                for (int i = 0; i < 8; ++i)
                    Bs[(nc + i) * LDSB + kk] = (_Float16)gb[i];
            }
        }
        if constexpr (AH) asm volatile("s_wait_asynccnt 0" ::: "memory");
        __syncthreads();

        #pragma unroll
        for (int ks = 0; ks < BK; ks += 32) {
            const v16h a0 = load_afrag(As, wm * 32,      ks, lrow, hi);
            const v16h a1 = load_afrag(As, wm * 32 + 16, ks, lrow, hi);
            const v16h b0 = load_bfrag(Bs, wn * 32,      ks, lrow, hi);
            const v16h b1 = load_bfrag(Bs, wn * 32 + 16, ks, lrow, hi);
            c00 = __builtin_amdgcn_wmma_f32_16x16x32_f16(false, a0, false, b0, (short)0, c00, false, false);
            c01 = __builtin_amdgcn_wmma_f32_16x16x32_f16(false, a0, false, b1, (short)0, c01, false, false);
            c10 = __builtin_amdgcn_wmma_f32_16x16x32_f16(false, a1, false, b0, (short)0, c10, false, false);
            c11 = __builtin_amdgcn_wmma_f32_16x16x32_f16(false, a1, false, b1, (short)0, c11, false, false);
        }
        __syncthreads();
    }

    const int row0 = m0 + wm * 32 + (hi ? 8 : 0);
    const int col0 = n0 + wn * 32 + lrow;
    auto st = [&](int row, int col, float val) {
        if (bias)     val += bias[col];
        if (residual) val += residual[(size_t)row * ldc + col];
        if (do_relu)  val = fmaxf(val, 0.0f);
        if constexpr (CH) C16[(size_t)row * ldc + col] = (_Float16)val;
        else              C  [(size_t)row * ldc + col] = val;
    };
    #pragma unroll
    for (int r = 0; r < 8; ++r) {
        st(row0 + r,      col0,      c00[r]);
        st(row0 + r,      col0 + 16, c01[r]);
        st(row0 + 16 + r, col0,      c10[r]);
        st(row0 + 16 + r, col0 + 16, c11[r]);
    }
}

// ============ logits GEMM: A f32 [M,K], B f16 [N,K] row-major ==============
__global__ __launch_bounds__(256)
void gemm_wmma_logits_kernel(const float* __restrict__ A,
                             const _Float16* __restrict__ Bf16,
                             const float* __restrict__ bias,
                             float* __restrict__ C, int M, int N, int K)
{
    __shared__ _Float16 As[64 * LDSA];
    __shared__ _Float16 Bs[128 * LDSB];

    const int tid  = threadIdx.x;
    const int lane = tid & 31;
    const int wave = tid >> 5;
    const int wm   = wave & 1;
    const int wn   = wave >> 1;
    const int m0   = blockIdx.y * 64;
    const int n0   = blockIdx.x * 128;
    const int lrow = lane & 15;
    const int hi   = lane >> 4;

    v8f c00 = {}, c01 = {}, c10 = {}, c11 = {};

    for (int k0 = 0; k0 < K; k0 += BK) {
        {
            const int r  = tid >> 2;
            const int kc = (tid & 3) * 16;
            const float* ga = A + (size_t)(m0 + r) * K + k0 + kc;
            _Float16* sa = As + r * LDSA + kc;
            #pragma unroll
            for (int i = 0; i < 16; ++i) sa[i] = (_Float16)ga[i];
        }
        {
            const int nn = tid >> 1;
            const int kc = (tid & 1) * 32;
            const _Float16* gb = Bf16 + (size_t)(n0 + nn) * K + k0 + kc;
            const unsigned int lds0 = (unsigned int)(uintptr_t)(&Bs[nn * LDSB + kc]);
            #pragma unroll
            for (int i = 0; i < 4; ++i) {
                asm volatile("global_load_async_to_lds_b128 %0, %1, off"
                             :: "v"(lds0 + 16u * i), "v"(gb + 8 * i) : "memory");
            }
        }
        asm volatile("s_wait_asynccnt 0" ::: "memory");
        __syncthreads();

        #pragma unroll
        for (int ks = 0; ks < BK; ks += 32) {
            const v16h a0 = load_afrag(As, wm * 32,      ks, lrow, hi);
            const v16h a1 = load_afrag(As, wm * 32 + 16, ks, lrow, hi);
            const v16h b0 = load_bfrag(Bs, wn * 32,      ks, lrow, hi);
            const v16h b1 = load_bfrag(Bs, wn * 32 + 16, ks, lrow, hi);
            c00 = __builtin_amdgcn_wmma_f32_16x16x32_f16(false, a0, false, b0, (short)0, c00, false, false);
            c01 = __builtin_amdgcn_wmma_f32_16x16x32_f16(false, a0, false, b1, (short)0, c01, false, false);
            c10 = __builtin_amdgcn_wmma_f32_16x16x32_f16(false, a1, false, b0, (short)0, c10, false, false);
            c11 = __builtin_amdgcn_wmma_f32_16x16x32_f16(false, a1, false, b1, (short)0, c11, false, false);
        }
        __syncthreads();
    }

    const int row0 = m0 + wm * 32 + (hi ? 8 : 0);
    const int col0 = n0 + wn * 32 + lrow;
    #pragma unroll
    for (int r = 0; r < 8; ++r) {
        C[(size_t)(row0 + r)      * N + col0]      = c00[r] + bias[col0];
        C[(size_t)(row0 + r)      * N + col0 + 16] = c01[r] + bias[col0 + 16];
        C[(size_t)(row0 + 16 + r) * N + col0]      = c10[r] + bias[col0];
        C[(size_t)(row0 + 16 + r) * N + col0 + 16] = c11[r] + bias[col0 + 16];
    }
}

// =================== f32 -> f16 conversion (embed table) ===================
__global__ __launch_bounds__(256)
void cvt_f32_f16_kernel(const float* __restrict__ in, _Float16* __restrict__ out,
                        int total)
{
    int idx = blockIdx.x * 256 + threadIdx.x;
    if (idx < total) out[idx] = (_Float16)in[idx];
}

// ====================== embedding + sinusoidal position ====================
__global__ __launch_bounds__(256)
void embed_pos_kernel(const int* __restrict__ tokens,
                      const float* __restrict__ embed,
                      float* __restrict__ out, int total)
{
    int idx = blockIdx.x * 256 + threadIdx.x;
    if (idx >= total) return;
    const int d  = idx % XD;
    const int rb = idx / XD;
    const int b  = rb % XB;
    const int p  = rb / XB;
    const int tok = tokens[p * XB + b];
    const float dim_scale = powf(10000.0f, 2.0f * (float)(d / 2) / (float)XD);
    const float phase = (d & 1) ? 1.57079632679f : 0.0f;
    out[idx] = embed[(size_t)tok * XD + d] * sqrtf((float)XD)
             + sinf((float)p / dim_scale + phase);
}

// ============================== layernorm ==================================
__global__ __launch_bounds__(256)
void layernorm_kernel(const float* __restrict__ X, const float* __restrict__ g,
                      const float* __restrict__ beta, float* __restrict__ Y,
                      int rows)
{
    const int lane = threadIdx.x & 31;
    const int wave = threadIdx.x >> 5;
    const int row  = blockIdx.x * 8 + wave;
    if (row >= rows) return;
    const float* x = X + (size_t)row * XD;
    float s = 0.0f;
    for (int i = lane; i < XD; i += 32) s += x[i];
    #pragma unroll
    for (int off = 16; off; off >>= 1) s += __shfl_xor(s, off, 32);
    const float mu = s / (float)XD;
    float v = 0.0f;
    for (int i = lane; i < XD; i += 32) { const float t = x[i] - mu; v += t * t; }
    #pragma unroll
    for (int off = 16; off; off >>= 1) v += __shfl_xor(v, off, 32);
    const float inv = rsqrtf(v / (float)XD + 1e-5f);
    float* y = Y + (size_t)row * XD;
    for (int i = lane; i < XD; i += 32) y[i] = (x[i] - mu) * inv * g[i] + beta[i];
}

// ==================== attention softmax (f16 scores, in place) =============
// S: [B*NH][m=512][n=512] f16.  wave per row, 8 rows per block.
// Applies 1/sqrt(embed_dim) scale, key mask, triu-style "causal" (n < m).
__global__ __launch_bounds__(256)
void attn_softmax_kernel(_Float16* __restrict__ S,
                         const unsigned char* __restrict__ key_mask, int causal)
{
    const int lane = threadIdx.x & 31;
    const int wave = threadIdx.x >> 5;
    const int rid  = blockIdx.x * 8 + wave;      // 0 .. 32*512-1
    const int batch = rid >> 9;                  // b*NH + h
    const int m     = rid & 511;
    const int b     = batch >> 3;
    _Float16* srow = S + (size_t)rid * XSEQ;
    const float scale = rsqrtf((float)XD);

    float vals[XSEQ / 32];
    float mx = -3.0e38f;
    #pragma unroll
    for (int j = 0; j < XSEQ / 32; ++j) {
        const int i = lane + j * 32;
        float p = (float)srow[i] * scale;
        if (key_mask && key_mask[i * XB + b]) p = NEG_INF;
        if (causal && i < m)                  p = NEG_INF;
        vals[j] = p; mx = fmaxf(mx, p);
    }
    #pragma unroll
    for (int off = 16; off; off >>= 1) mx = fmaxf(mx, __shfl_xor(mx, off, 32));
    float sum = 0.0f;
    #pragma unroll
    for (int j = 0; j < XSEQ / 32; ++j) {
        const float e = __expf(vals[j] - mx);
        vals[j] = e; sum += e;
    }
    #pragma unroll
    for (int off = 16; off; off >>= 1) sum += __shfl_xor(sum, off, 32);
    const float inv = 1.0f / sum;
    #pragma unroll
    for (int j = 0; j < XSEQ / 32; ++j)
        srow[lane + j * 32] = (_Float16)(vals[j] * inv);
}

// ============================ log-softmax (in place) =======================
__global__ __launch_bounds__(256)
void logsoftmax_kernel(float* __restrict__ X, int Vn)
{
    __shared__ float red[256];
    const int row = blockIdx.x, tid = threadIdx.x;
    float* x = X + (size_t)row * Vn;
    float lmax = -3.0e38f;
    for (int i = tid; i < Vn; i += 256) lmax = fmaxf(lmax, x[i]);
    red[tid] = lmax; __syncthreads();
    for (int s = 128; s > 0; s >>= 1) {
        if (tid < s) red[tid] = fmaxf(red[tid], red[tid + s]);
        __syncthreads();
    }
    const float mx = red[0]; __syncthreads();
    float lsum = 0.0f;
    for (int i = tid; i < Vn; i += 256) lsum += __expf(x[i] - mx);
    red[tid] = lsum; __syncthreads();
    for (int s = 128; s > 0; s >>= 1) {
        if (tid < s) red[tid] += red[tid + s];
        __syncthreads();
    }
    const float lse = mx + __logf(red[0]);
    for (int i = tid; i < Vn; i += 256) x[i] = x[i] - lse;
}

// ============================== orchestration ==============================
extern "C" void kernel_launch(void* const* d_in, const int* in_sizes, int n_in,
                              void* d_out, int out_size, void* d_ws, size_t ws_size,
                              hipStream_t stream)
{
    const int* src_tokens = (const int*)d_in[0];
    const int* tgt_tokens = (const int*)d_in[1];
    const unsigned char* src_mask = (const unsigned char*)d_in[2];
    const unsigned char* tgt_mask = (const unsigned char*)d_in[3];
    const float* P[64];
    for (int i = 4; i < n_in && i < 64; ++i) P[i] = (const float*)d_in[i];
    // params flattened in setup_inputs() dict insertion order
    const float* embed    = P[4];
    const float* logits_b = P[5];
    const float* e_ln1g = P[6],  *e_ln1b = P[7],  *e_ln2g = P[8],  *e_ln2b = P[9];
    const float* e_W1 = P[10], *e_b1 = P[11], *e_W2 = P[12], *e_b2 = P[13];
    const float* e_Wq = P[14], *e_bq = P[15], *e_Wk = P[16], *e_bk = P[17];
    const float* e_Wv = P[18], *e_bv = P[19], *e_Wo = P[20], *e_bo = P[21];
    const float* d_ln1g = P[22], *d_ln1b = P[23], *d_ln2g = P[24], *d_ln2b = P[25];
    const float* d_ln3g = P[26], *d_ln3b = P[27];
    const float* d_W1 = P[28], *d_b1 = P[29], *d_W2 = P[30], *d_b2 = P[31];
    const float* sWq = P[32], *sbq = P[33], *sWk = P[34], *sbk = P[35];
    const float* sWv = P[36], *sbv = P[37], *sWo = P[38], *sbo = P[39];
    const float* eWq = P[40], *ebq = P[41], *eWk = P[42], *ebk = P[43];
    const float* eWv = P[44], *ebv = P[45], *eWo = P[46], *ebo = P[47];
    const float* enc_ln_g = P[48], *enc_ln_b = P[49];
    const float* dec_ln_g = P[50], *dec_ln_b = P[51];
    const float* out_W = P[52], *out_b = P[53];

    // ---- workspace: 9*TD f32 acts + V*D f16 embed + 32*512*512 f16 scores --
    const size_t TD = (size_t)XT * XD;
    float* ws     = (float*)d_ws;
    float* x      = ws + 0 * TD;
    float* xn     = ws + 1 * TD;
    float* q      = ws + 2 * TD;
    float* k      = ws + 3 * TD;
    float* v      = ws + 4 * TD;
    float* pooled = ws + 5 * TD;
    float* encout = ws + 6 * TD;
    float* h      = ws + 7 * TD;
    float* hid    = ws + 2 * TD;                       // overlaps q/k/v/pooled in FFN
    _Float16* embedF16 = (_Float16*)(ws + 9 * TD);     // 16.38M halfs
    _Float16* S16      = (_Float16*)(ws + 9 * TD + ((size_t)XV * XD) / 2);

    auto gemm = [&](const float* A, const float* B, const float* bias,
                    const float* res, float* C, int M, int N, int K, int relu) {
        dim3 g(N / 128, M / 64, 1);
        gemm2_kernel<64, 128, 0, 0, 0><<<g, 256, 0, stream>>>(
            A, B, bias, res, C, M, N, K, K, N, N, 0, 0, 0, relu);
    };
    auto lnorm = [&](const float* X, const float* g, const float* b, float* Y) {
        layernorm_kernel<<<XT / 8, 256, 0, stream>>>(X, g, b, Y, XT);
    };
    // WMMA attention: scores GEMM -> masked softmax -> prob@V GEMM
    auto attn = [&](const float* Q, const float* K_, const float* V_,
                    const unsigned char* mask, float* O, int causal) {
        const long sQK = 64, sS = (long)XSEQ * XSEQ;
        {   // S[bh,m,n] = Q . K^T   (A f32, B "transposed", C f16)
            dim3 g(XSEQ / 128, XSEQ / 64, XB * XNH);
            gemm2_kernel<64, 128, 1, 0, 1><<<g, 256, 0, stream>>>(
                Q, K_, nullptr, nullptr, (void*)S16,
                XSEQ, XSEQ, XHD, XB * XD, XB * XD, XSEQ, sQK, sQK, sS, 0);
        }
        attn_softmax_kernel<<<(XB * XNH * XSEQ) / 8, 256, 0, stream>>>(S16, mask, causal);
        {   // O[m, b, h*64+d] = P . V  (A f16 via async staging)
            dim3 g(XHD / 64, XSEQ / 128, XB * XNH);
            gemm2_kernel<128, 64, 0, 1, 0><<<g, 256, 0, stream>>>(
                (const void*)S16, V_, nullptr, nullptr, O,
                XSEQ, XHD, XSEQ, XSEQ, XB * XD, XB * XD, sS, sQK, sQK, 0);
        }
    };

    const int DD = XD * XD, DHs = XD * XH;

    cvt_f32_f16_kernel<<<(XV * XD + 255) / 256, 256, 0, stream>>>(embed, embedF16, XV * XD);

    // =========================== encoder ===================================
    embed_pos_kernel<<<(XT * XD + 255) / 256, 256, 0, stream>>>(src_tokens, embed, x, XT * XD);
    for (int l = 0; l < XL; ++l) {
        lnorm(x, e_ln1g + l * XD, e_ln1b + l * XD, xn);
        gemm(xn, e_Wq + (size_t)l * DD, e_bq + l * XD, nullptr, q, XT, XD, XD, 0);
        gemm(xn, e_Wk + (size_t)l * DD, e_bk + l * XD, nullptr, k, XT, XD, XD, 0);
        gemm(xn, e_Wv + (size_t)l * DD, e_bv + l * XD, nullptr, v, XT, XD, XD, 0);
        attn(q, k, v, src_mask, pooled, 0);
        gemm(pooled, e_Wo + (size_t)l * DD, e_bo + l * XD, x, x, XT, XD, XD, 0);
        lnorm(x, e_ln2g + l * XD, e_ln2b + l * XD, xn);
        gemm(xn, e_W1 + (size_t)l * DHs, e_b1 + l * XH, nullptr, hid, XT, XH, XD, 1);
        gemm(hid, e_W2 + (size_t)l * DHs, e_b2 + l * XD, x, x, XT, XD, XH, 0);
    }
    lnorm(x, enc_ln_g, enc_ln_b, encout);

    // =========================== decoder ===================================
    embed_pos_kernel<<<(XT * XD + 255) / 256, 256, 0, stream>>>(tgt_tokens, embed, h, XT * XD);
    for (int l = 0; l < XL; ++l) {
        lnorm(h, d_ln1g + l * XD, d_ln1b + l * XD, xn);
        gemm(xn, sWq + (size_t)l * DD, sbq + l * XD, nullptr, q, XT, XD, XD, 0);
        gemm(xn, sWk + (size_t)l * DD, sbk + l * XD, nullptr, k, XT, XD, XD, 0);
        gemm(xn, sWv + (size_t)l * DD, sbv + l * XD, nullptr, v, XT, XD, XD, 0);
        attn(q, k, v, tgt_mask, pooled, 1);
        gemm(pooled, sWo + (size_t)l * DD, sbo + l * XD, h, h, XT, XD, XD, 0);
        lnorm(h, d_ln2g + l * XD, d_ln2b + l * XD, xn);
        gemm(xn,     eWq + (size_t)l * DD, ebq + l * XD, nullptr, q, XT, XD, XD, 0);
        gemm(encout, eWk + (size_t)l * DD, ebk + l * XD, nullptr, k, XT, XD, XD, 0);
        gemm(encout, eWv + (size_t)l * DD, ebv + l * XD, nullptr, v, XT, XD, XD, 0);
        attn(q, k, v, src_mask, pooled, 0);
        gemm(pooled, eWo + (size_t)l * DD, ebo + l * XD, h, h, XT, XD, XD, 0);
        lnorm(h, d_ln3g + l * XD, d_ln3b + l * XD, xn);
        gemm(xn, d_W1 + (size_t)l * DHs, d_b1 + l * XH, nullptr, hid, XT, XH, XD, 1);
        gemm(hid, d_W2 + (size_t)l * DHs, d_b2 + l * XD, h, h, XT, XD, XH, 0);
    }
    lnorm(h, dec_ln_g, dec_ln_b, xn);
    gemm(xn, out_W, out_b, nullptr, x, XT, XD, XD, 0);

    // logits = x @ embedF16.T + logits_b -> d_out, then in-place log_softmax
    float* logits = (float*)d_out;
    {
        dim3 g(XV / 128, XT / 64);
        gemm_wmma_logits_kernel<<<g, 256, 0, stream>>>(x, embedF16, logits_b,
                                                       logits, XT, XV, XD);
    }
    logsoftmax_kernel<<<XT, 256, 0, stream>>>(logits, XV);
}
